// LSTMModel_71932112273507
// MI455X (gfx1250) — matrix operand order repaired and checked by reading, not compiled, not run on hardware
//
#include <hip/hip_runtime.h>

// ---------------------------------------------------------------------------
// LSTM: B=32, T=512, D=1024, H=1024  (Keras gate order i,f,g,o)
//   Prep : x -> bf16, W -> bf16 transposed (n-major), U -> bf16 transposed
//   Phase 1: xz[T][B][4H] = x @ W + b   (bf16 WMMA GEMM, double-buffered LDS,
//            async global->LDS staging when the builtin is available)
//   Phase 2: persistent scan kernel, 16 WGs, grid barrier per timestep
// ---------------------------------------------------------------------------

typedef __attribute__((ext_vector_type(16))) __bf16 bf16x16;
typedef __attribute__((ext_vector_type(8)))  float  f32x8;
typedef __attribute__((ext_vector_type(4)))  int    i32x4;

union Frag32B { uint4 u[2]; bf16x16 v; };

#define BB 32
#define TT 512
#define DD 1024
#define HH 1024
#define NG 4096              // 4*H

// workspace layout (bytes)
#define XZ_BYTES  ((size_t)TT * BB * NG * 4)     // 256 MiB  f32  [T][B][4H]
#define WT_BYTES  ((size_t)NG * DD * 2)          //   8 MiB  bf16 [4H][D] n-major
#define UT_BYTES  ((size_t)NG * HH * 2)          //   8 MiB  bf16 [4H][H] n-major
#define XBF_BYTES ((size_t)BB * TT * DD * 2)     //  32 MiB  bf16 x
#define HBF_BYTES ((size_t)2 * BB * HH * 2)      // 128 KiB  bf16 h, double buffered
#define C_BYTES   ((size_t)BB * HH * 4)          // 128 KiB  f32  c

#define NWG_SCAN 16

// ---------------------------------------------------------------------------
// async global->LDS copy (16B per lane) with graceful fallback
// ---------------------------------------------------------------------------
typedef __attribute__((address_space(1))) i32x4 glob_i32x4;
typedef __attribute__((address_space(3))) i32x4 lds_i32x4;

#if defined(__has_builtin)
#  if __has_builtin(__builtin_amdgcn_global_load_async_to_lds_b128)
#    define LSTM_HAVE_ASYNC_LDS 1
#  endif
#endif

__device__ __forceinline__ void async_cp16(const void* g, void* l) {
#ifdef LSTM_HAVE_ASYNC_LDS
    __builtin_amdgcn_global_load_async_to_lds_b128((glob_i32x4*)g, (lds_i32x4*)l, 0, 0);
#else
    *(uint4*)l = *(const uint4*)g;
#endif
}

__device__ __forceinline__ void wait_async() {
#ifdef LSTM_HAVE_ASYNC_LDS
#  if __has_builtin(__builtin_amdgcn_s_wait_asynccnt)
    __builtin_amdgcn_s_wait_asynccnt(0);
#  else
    asm volatile("s_wait_asynccnt 0" ::: "memory");
#  endif
#endif
}

__device__ __forceinline__ float fsig(float x) {
    return 1.f / (1.f + __expf(-x));
}
__device__ __forceinline__ float ftanh(float x) {
    float ax = fabsf(x);
    float e  = __expf(-2.f * ax);
    float r  = (1.f - e) / (1.f + e);
    return copysignf(r, x);
}

// ---------------------------------------------------------------------------
// Prep A: dst[n][k] = bf16(src[k][n]) for 1024x4096 matrices (W and U)
// ---------------------------------------------------------------------------
__global__ __launch_bounds__(256) void lstm_prep_tr(const float* __restrict__ src,
                                                    __bf16* __restrict__ dst) {
    size_t i = (size_t)blockIdx.x * 256 + threadIdx.x;   // 0 .. 4M-1, coalesced reads
    int n = (int)(i & (NG - 1));
    int k = (int)(i >> 12);
    dst[(size_t)n * 1024 + k] = (__bf16)src[i];
}

// ---------------------------------------------------------------------------
// Prep B: straight f32 -> bf16 conversion (x), 4 elems/thread
// ---------------------------------------------------------------------------
__global__ __launch_bounds__(256) void lstm_prep_cvt(const float* __restrict__ src,
                                                     __bf16* __restrict__ dst) {
    size_t i = ((size_t)blockIdx.x * 256 + threadIdx.x) * 4;
    const float4 f = *(const float4*)&src[i];
    union { __bf16 h[4]; uint2 u; } p;
    p.h[0] = (__bf16)f.x;  p.h[1] = (__bf16)f.y;
    p.h[2] = (__bf16)f.z;  p.h[3] = (__bf16)f.w;
    *(uint2*)&dst[i] = p.u;
}

// ---------------------------------------------------------------------------
// Prep C: zero c, zero both h bf16 buffers, reset grid barrier
// ---------------------------------------------------------------------------
__global__ __launch_bounds__(256) void lstm_prep_state(float* __restrict__ cbuf,
                                                       __bf16* __restrict__ hbf,
                                                       unsigned* __restrict__ bar) {
    int i = blockIdx.x * 256 + threadIdx.x;              // 0 .. 65535
    if (i < BB * HH) cbuf[i] = 0.f;
    hbf[i] = (__bf16)0.f;                                // 2*32*1024 elems
    if (i < 4) bar[i] = 0u;
}

// ---------------------------------------------------------------------------
// Phase 1: xz = x @ W + b,  128x128 tile, K-step 32, bf16 WMMA,
//   double-buffered LDS staging (async global->LDS when available).
//   grid (128, 32) blocks of 256 threads (8 waves: 2 along M x 4 along N)
// ---------------------------------------------------------------------------
__global__ __launch_bounds__(256) void lstm_xz_gemm(const __bf16* __restrict__ xbf,
                                                    const __bf16* __restrict__ Wt,
                                                    const float* __restrict__ bias,
                                                    float* __restrict__ xz) {
    const int tid  = threadIdx.x;
    const int wid  = tid >> 5;
    const int lane = tid & 31;
    const int lrow = lane & 15, lhi = lane >> 4;
    const int m0 = blockIdx.x * 128;
    const int n0 = blockIdx.y * 128;
    const int wm = wid & 1;          // 64-row half
    const int wn = wid >> 1;         // 32-col quarter

    __shared__ __bf16 As[2][128][40];   // [m][k], row stride 80B (16B aligned)
    __shared__ __bf16 Bs[2][128][40];   // [n][k] (Wt is already n-major)

    // stage one K-tile (128 rows x 32 k of both A and B) = 4x 16B copies/thread
    auto stage = [&](int buf, int k0) {
        for (int j = 0; j < 2; ++j) {
            int c  = tid + 256 * j;          // 512 chunks, 4 per row
            int r  = c >> 2;
            int kc = (c & 3) << 3;
            async_cp16(&xbf[(size_t)(m0 + r) * DD + k0 + kc], &As[buf][r][kc]);
            async_cp16(&Wt [(size_t)(n0 + r) * DD + k0 + kc], &Bs[buf][r][kc]);
        }
    };

    f32x8 acc[4][2] = {};

    stage(0, 0);
    wait_async();
    __syncthreads();

    for (int kt = 0; kt < 32; ++kt) {
        const int buf = kt & 1;
        if (kt + 1 < 32) stage(buf ^ 1, (kt + 1) * 32);

        Frag32B a[4], b[2];
        for (int mt = 0; mt < 4; ++mt) {
            const __bf16* p = &As[buf][64 * wm + 16 * mt + lrow][8 * lhi];
            a[mt].u[0] = *(const uint4*)p;
            a[mt].u[1] = *(const uint4*)(p + 16);   // K += 16
        }
        for (int nt = 0; nt < 2; ++nt) {
            const __bf16* p = &Bs[buf][32 * wn + 16 * nt + lrow][8 * lhi];
            b[nt].u[0] = *(const uint4*)p;
            b[nt].u[1] = *(const uint4*)(p + 16);
        }
        for (int mt = 0; mt < 4; ++mt)
            for (int nt = 0; nt < 2; ++nt)
                acc[mt][nt] = __builtin_amdgcn_wmma_f32_16x16x32_bf16(
                    false, a[mt].v, false, b[nt].v, (short)0, acc[mt][nt],
                    false, false);

        wait_async();
        __syncthreads();
    }

    // epilogue: add bias, scatter rows m=b*512+t into xz[t][b][n]
    for (int nt = 0; nt < 2; ++nt) {
        const int n  = n0 + 32 * wn + 16 * nt + lrow;
        const float bb = bias[n];
        for (int mt = 0; mt < 4; ++mt) {
            for (int r = 0; r < 8; ++r) {
                int m  = m0 + 64 * wm + 16 * mt + r + 8 * lhi;
                int t  = m & (TT - 1);
                int bi = m >> 9;
                xz[((size_t)t * BB + bi) * NG + n] = acc[mt][nt][r] + bb;
            }
        }
    }
}

// ---------------------------------------------------------------------------
// grid barrier: sense via generation counter, L2 atomics, s_sleep spin
// ---------------------------------------------------------------------------
__device__ __forceinline__ void grid_barrier(unsigned* bar, int nwg) {
    __syncthreads();
    __threadfence();
    if (threadIdx.x == 0) {
        unsigned gen = atomicAdd(&bar[1], 0u);
        if (atomicAdd(&bar[0], 1u) == (unsigned)(nwg - 1)) {
            atomicExch(&bar[0], 0u);
            __threadfence();
            atomicAdd(&bar[1], 1u);
        } else {
            while (atomicAdd(&bar[1], 0u) == gen)
                __builtin_amdgcn_s_sleep(2);
        }
    }
    __threadfence();
    __syncthreads();
}

// ---------------------------------------------------------------------------
// Phase 2: persistent scan. 16 WGs x 256 threads. WG w owns h cols
// [64w, 64w+64). Waves: gate = wid>>1 (i,f,g,o), half = wid&1 (32 cols).
// Per step: acc = h_bf16 @ Ut (WMMA), z = acc + xz[t] -> LDS, gates,
// update c/h, publish h bf16 (double buffered), grid barrier.
// ---------------------------------------------------------------------------
__global__ __launch_bounds__(256) void lstm_scan(const float* __restrict__ xz,
                                                 const __bf16* __restrict__ Ut,
                                                 float* __restrict__ out,
                                                 __bf16* __restrict__ hbf,
                                                 float* __restrict__ cbuf,
                                                 unsigned* __restrict__ bar) {
    const int tid  = threadIdx.x;
    const int wid  = tid >> 5;
    const int lane = tid & 31;
    const int lrow = lane & 15, lhi = lane >> 4;
    const int jb   = blockIdx.x * 64;          // owned h columns
    const int gate = wid >> 1;                 // 0..3
    const int half = wid & 1;                  // 0..1
    const int nbase = gate * HH + jb + 32 * half;   // z-column base for wave

    __shared__ float zbuf[4][BB][64];          // 32 KiB gate exchange

    for (int t = 0; t < TT; ++t) {
        const __bf16* hsrc = hbf + (size_t)(t & 1) * (BB * HH);
        const float*  xzt  = xz + (size_t)t * BB * NG;

        // warm L2 with this step's xz slice while the WMMA loop runs
        {
            int ch = tid >> 1;                 // 128 chunks = 32 rows x 4 gates
            int br = ch & 31, g = ch >> 5;
            __builtin_prefetch(xzt + (size_t)br * NG + g * HH + jb + ((tid & 1) << 5), 0, 0);
        }

        f32x8 acc[2][2] = {};
        for (int ks = 0; ks < 32; ++ks) {
            Frag32B a[2], b[2];
            for (int mt = 0; mt < 2; ++mt) {
                const __bf16* p = hsrc + (size_t)(16 * mt + lrow) * HH + 32 * ks + 8 * lhi;
                a[mt].u[0] = *(const uint4*)p;
                a[mt].u[1] = *(const uint4*)(p + 16);
            }
            for (int nt = 0; nt < 2; ++nt) {
                const __bf16* p = Ut + (size_t)(nbase + 16 * nt + lrow) * HH + 32 * ks + 8 * lhi;
                b[nt].u[0] = *(const uint4*)p;
                b[nt].u[1] = *(const uint4*)(p + 16);
            }
            for (int mt = 0; mt < 2; ++mt)
                for (int nt = 0; nt < 2; ++nt)
                    acc[mt][nt] = __builtin_amdgcn_wmma_f32_16x16x32_bf16(
                        false, a[mt].v, false, b[nt].v, (short)0, acc[mt][nt],
                        false, false);
        }

        // z = h@U + xz (bias already folded into xz) -> LDS
        for (int nt = 0; nt < 2; ++nt) {
            const int n  = nbase + 16 * nt + lrow;
            const int jl = 32 * half + 16 * nt + lrow;
            for (int mt = 0; mt < 2; ++mt)
                for (int r = 0; r < 8; ++r) {
                    int brow = 16 * mt + r + 8 * lhi;
                    zbuf[gate][brow][jl] = acc[mt][nt][r] + xzt[(size_t)brow * NG + n];
                }
        }
        __syncthreads();

        // elementwise gate math; publish h into the other bf16 buffer
        __bf16* hdst = hbf + (size_t)((t + 1) & 1) * (BB * HH);
        for (int e = 0; e < 8; ++e) {
            int cell = tid + 256 * e;          // 2048 cells = 32 rows x 64 cols
            int brow = cell >> 6, j = cell & 63;
            float iv = fsig(zbuf[0][brow][j]);
            float fv = fsig(zbuf[1][brow][j]);
            float gv = ftanh(zbuf[2][brow][j]);
            float ov = fsig(zbuf[3][brow][j]);
            size_t cidx = (size_t)brow * HH + jb + j;
            float cn = fv * cbuf[cidx] + iv * gv;
            float hn = ov * ftanh(cn);
            cbuf[cidx] = cn;
            out[(size_t)brow * TT * HH + (size_t)t * HH + jb + j] = hn;
            hdst[cidx] = (__bf16)hn;
        }

        grid_barrier(bar, NWG_SCAN);
    }
}

// ---------------------------------------------------------------------------
extern "C" void kernel_launch(void* const* d_in, const int* in_sizes, int n_in,
                              void* d_out, int out_size, void* d_ws, size_t ws_size,
                              hipStream_t stream) {
    const float* x = (const float*)d_in[0];   // [32,512,1024]
    const float* W = (const float*)d_in[1];   // [1024,4096]
    const float* U = (const float*)d_in[2];   // [1024,4096]
    const float* b = (const float*)d_in[3];   // [4096]
    float* out = (float*)d_out;               // [32,512,1024]

    char* ws = (char*)d_ws;
    size_t off = 0;
    float*    xz   = (float*)(ws + off);           off += XZ_BYTES;
    __bf16*   Wt   = (__bf16*)(ws + off);          off += WT_BYTES;
    __bf16*   Ut   = (__bf16*)(ws + off);          off += UT_BYTES;
    __bf16*   xbf  = (__bf16*)(ws + off);          off += XBF_BYTES;
    __bf16*   hbf  = (__bf16*)(ws + off);          off += HBF_BYTES;
    float*    cbuf = (float*)(ws + off);           off += C_BYTES;
    unsigned* bar  = (unsigned*)(ws + off);

    lstm_prep_tr<<<(NG * DD) / 256, 256, 0, stream>>>(W, Wt);
    lstm_prep_tr<<<(NG * HH) / 256, 256, 0, stream>>>(U, Ut);
    lstm_prep_cvt<<<(BB * TT * DD) / 1024, 256, 0, stream>>>(x, xbf);
    lstm_prep_state<<<256, 256, 0, stream>>>(cbuf, hbf, bar);
    lstm_xz_gemm<<<dim3((BB * TT) / 128, NG / 128), 256, 0, stream>>>(xbf, Wt, b, xz);
    lstm_scan<<<NWG_SCAN, 256, 0, stream>>>(xz, Ut, out, hbf, cbuf, bar);
}
